// SGL_13159779795285
// MI455X (gfx1250) — matrix operand order repaired and checked by reading, not compiled, not run on hardware
//
#include <hip/hip_runtime.h>
#include <hip/hip_bf16.h>
#include <hip/hip_fp16.h>

// ---------------------------------------------------------------------------
// SGL loss on MI455X (gfx1250).
//  * SpMM layers: bandwidth-bound scatter-add (atomics), float4 vectorized,
//    with global_prefetch_b8 for the next gathered row.
//  * InfoNCE logits GEMM: v_wmma_f32_16x16x32_f16 (wave32 WMMA), fused online
//    sum-of-exp per row (dots/T in [-5,5] -> no max pass needed).
//  * Workspace: 2 full-node layer buffers (2x64MB) + 7 batch mats + 4 f16
//    normalized mats  ~= 138 MB.
// ---------------------------------------------------------------------------

#define NUSERS  150000
#define NITEMS  100000
#define NNODES  250000
#define DEMB    64
#define NEDGE   4000000
#define BATCH   4096
#define TEMPER  0.2f
#define EMB_REG 2.5e-5f
#define SSL_REG 0.1f

typedef __attribute__((ext_vector_type(16))) _Float16 v16h;
typedef __attribute__((ext_vector_type(8)))  _Float16 v8h;
typedef __attribute__((ext_vector_type(8)))  float    v8f;

// ---------------- SpMM: out[h[e]] += vals[e] * in[t[e]]  (scatter, atomics) --
// Layer 1 reads the virtual concat [user_emb ; item_emb] via (inA, inB, split).
__global__ void sgl_spmm_kernel(const int* __restrict__ h, const int* __restrict__ t,
                                const float* __restrict__ vals,
                                const float* __restrict__ inA, const float* __restrict__ inB,
                                int split, float* __restrict__ out)
{
    const long total  = (long)NEDGE * 16;           // 16 lanes (float4 each) per edge
    const long stride = (long)gridDim.x * blockDim.x;
    for (long i = (long)blockIdx.x * blockDim.x + threadIdx.x; i < total; i += stride) {
        int e   = (int)(i >> 4);
        int seg = (int)(i & 15);
        // prefetch the row this thread will gather next iteration
        long ni = i + stride;
        if (ni < total) {
            int nt = t[(int)(ni >> 4)];
            const float* np = (nt < split) ? (inA + (long)nt * DEMB)
                                           : (inB + (long)(nt - split) * DEMB);
            __builtin_prefetch(np, 0, 1);
        }
        float v = vals[e];
        if (v == 0.0f) continue;                    // dropout-zeroed edges
        int tt = t[e];
        const float* src = (tt < split) ? (inA + (long)tt * DEMB)
                                        : (inB + (long)(tt - split) * DEMB);
        float4 s = *(const float4*)(src + seg * 4);
        float* dst = out + (long)h[e] * DEMB + seg * 4;
        atomicAdd(dst + 0, v * s.x);
        atomicAdd(dst + 1, v * s.y);
        atomicAdd(dst + 2, v * s.z);
        atomicAdd(dst + 3, v * s.w);
    }
}

// ------------- extract: out[b] = e0[idx[b]] + L1[idx[b]+off] + L2[idx[b]+off]
__global__ void sgl_extract_kernel(const float* __restrict__ base,
                                   const float* __restrict__ l1,
                                   const float* __restrict__ l2,
                                   const int* __restrict__ idx, int node_off,
                                   float* __restrict__ out)
{
    int i = blockIdx.x * blockDim.x + threadIdx.x;  // BATCH*16 float4 slots
    if (i >= BATCH * 16) return;
    int b = i >> 4, seg = i & 15;
    int id = idx[b];
    long r0 = (long)id * DEMB + seg * 4;
    long r1 = (long)(id + node_off) * DEMB + seg * 4;
    float4 a = *(const float4*)(base + r0);
    float4 x = *(const float4*)(l1 + r1);
    float4 y = *(const float4*)(l2 + r1);
    float4 o;
    o.x = a.x + x.x + y.x;  o.y = a.y + x.y + y.y;
    o.z = a.z + x.z + y.z;  o.w = a.w + x.w + y.w;
    *(float4*)(out + (long)b * DEMB + seg * 4) = o;
}

// ------------- row l2-normalize + convert to f16 ---------------------------
__global__ void sgl_norm_f16_kernel(const float* __restrict__ in, _Float16* __restrict__ out)
{
    int b = blockIdx.x * blockDim.x + threadIdx.x;
    if (b >= BATCH) return;
    const float* r = in + (long)b * DEMB;
    float ss = 0.f;
    #pragma unroll
    for (int j = 0; j < DEMB; ++j) ss += r[j] * r[j];
    float inv = 1.0f / fmaxf(sqrtf(ss), 1e-12f);
    _Float16* o = out + (long)b * DEMB;
    #pragma unroll
    for (int j = 0; j < DEMB; ++j) o[j] = (_Float16)(r[j] * inv);
}

// ------------- BPR: mean(softplus(u.neg - u.pos)) --------------------------
__global__ void sgl_bpr_kernel(const float* __restrict__ u, const float* __restrict__ p,
                               const float* __restrict__ n, float* __restrict__ out)
{
    int b = blockIdx.x * blockDim.x + threadIdx.x;
    if (b >= BATCH) return;
    float dp = 0.f, dn = 0.f;
    #pragma unroll
    for (int j = 0; j < DEMB; ++j) {
        float uv = u[(long)b * DEMB + j];
        dp += uv * p[(long)b * DEMB + j];
        dn += uv * n[(long)b * DEMB + j];
    }
    float x  = dn - dp;
    float sp = (x > 20.f) ? x : log1pf(__expf(x));
    atomicAdd(out + 0, sp * (1.0f / BATCH));
}

// ------------- embedding L2 regularizer ------------------------------------
__global__ void sgl_emb_kernel(const float* __restrict__ ue, const float* __restrict__ ie,
                               const int* __restrict__ us, const int* __restrict__ ps,
                               const int* __restrict__ ns, float* __restrict__ out)
{
    int b = blockIdx.x * blockDim.x + threadIdx.x;
    if (b >= BATCH) return;
    const float* r0 = ue + (long)us[b] * DEMB;
    const float* r1 = ie + (long)ps[b] * DEMB;
    const float* r2 = ie + (long)ns[b] * DEMB;
    float s = 0.f;
    #pragma unroll
    for (int j = 0; j < DEMB; ++j)
        s += r0[j] * r0[j] + r1[j] * r1[j] + r2[j] * r2[j];
    atomicAdd(out + 3, EMB_REG * s);
}

// ------------- WMMA operand loaders (ISA 7.12.2 16-bit layouts) ------------
// A (16x32, MxK): lane = row (lane&15); lanes 0-15 hold K [k0..k0+7],[k0+16..k0+23],
//                 lanes 16-31 hold K [k0+8..k0+15],[k0+24..k0+31].
__device__ __forceinline__ v16h sgl_load_a(const _Float16* __restrict__ mat,
                                           int lane, int row0, int k0)
{
    int r  = row0 + (lane & 15);
    int hi = (lane >> 4) & 1;
    const _Float16* p = mat + (size_t)r * DEMB + k0 + hi * 8;
    v8h lo = *(const v8h*)p;
    v8h up = *(const v8h*)(p + 16);
    v16h o;
    #pragma unroll
    for (int i = 0; i < 8; ++i) { o[i] = lo[i]; o[8 + i] = up[i]; }
    return o;
}
// B (32x16, KxN): lane&15 = column n (== a row of X2, since logits = X1*X2^T);
//                 lanes 0-15 hold K [k0..k0+7],[k0+16..k0+23], lanes 16-31 hold K [k0+8..k0+15],[k0+24..k0+31].
__device__ __forceinline__ v16h sgl_load_b(const _Float16* __restrict__ mat,
                                           int lane, int col0, int k0)
{
    int c  = col0 + (lane & 15);
    int hi = (lane >> 4) & 1;
    const _Float16* p = mat + (size_t)c * DEMB + k0 + hi * 8;
    v8h lo = *(const v8h*)p;
    v8h up = *(const v8h*)(p + 16);
    v16h o;
    #pragma unroll
    for (int i = 0; i < 8; ++i) { o[i] = lo[i]; o[8 + i] = up[i]; }
    return o;
}

// ------------- InfoNCE denominator: per row, log(sum_j exp(x1_i.x2_j / T)) --
// One wave owns a 16-row tile; streams 256 16-col tiles; 2 WMMA (K=32) each.
// C layout: lanes 0-15 -> (M=vgpr, N=lane); lanes 16-31 -> (M=vgpr+8, N=lane-16).
__global__ void sgl_nce_gemm_kernel(const _Float16* __restrict__ X1,
                                    const _Float16* __restrict__ X2,
                                    float* __restrict__ out, float scale)
{
    int wave = threadIdx.x >> 5;
    int lane = threadIdx.x & 31;
    int rt   = blockIdx.x * (blockDim.x >> 5) + wave;   // row tile 0..255
    int row0 = rt * 16;

    v16h a0 = sgl_load_a(X1, lane, row0, 0);
    v16h a1 = sgl_load_a(X1, lane, row0, 32);

    float se[8];
    #pragma unroll
    for (int v = 0; v < 8; ++v) se[v] = 0.f;

    const float invT = 1.0f / TEMPER;
    for (int jt = 0; jt < BATCH / 16; ++jt) {
        if (jt + 1 < BATCH / 16)
            __builtin_prefetch(X2 + (size_t)(jt + 1) * 16 * DEMB, 0, 1);
        v16h b0 = sgl_load_b(X2, lane, jt * 16, 0);
        v16h b1 = sgl_load_b(X2, lane, jt * 16, 32);
        v8f c = {};
        c = __builtin_amdgcn_wmma_f32_16x16x32_f16(false, a0, false, b0,
                                                   (short)0, c, false, false);
        c = __builtin_amdgcn_wmma_f32_16x16x32_f16(false, a1, false, b1,
                                                   (short)0, c, false, false);
        #pragma unroll
        for (int v = 0; v < 8; ++v) se[v] += __expf(c[v] * invT);
    }

    // reduce each row's partial sums across its 16-lane half
    #pragma unroll
    for (int v = 0; v < 8; ++v) {
        float s = se[v];
        #pragma unroll
        for (int m = 1; m < 16; m <<= 1) s += __shfl_xor(s, m, 16);
        se[v] = s;
    }
    if ((lane & 15) == 0) {   // lane 0: rows row0+0..7 ; lane 16: rows row0+8..15
        float acc = 0.f;
        #pragma unroll
        for (int v = 0; v < 8; ++v) acc += __logf(se[v]);
        atomicAdd(out, scale * acc);
    }
}

// ------------- InfoNCE numerator: -mean(x1_b . x2_b)/T ---------------------
__global__ void sgl_nce_pos_kernel(const _Float16* __restrict__ X1,
                                   const _Float16* __restrict__ X2,
                                   float* __restrict__ out, float scale)
{
    int b = blockIdx.x * blockDim.x + threadIdx.x;
    if (b >= BATCH) return;
    float d = 0.f;
    #pragma unroll
    for (int j = 0; j < DEMB; ++j)
        d += (float)X1[(long)b * DEMB + j] * (float)X2[(long)b * DEMB + j];
    atomicAdd(out, scale * d);
}

// ---------------------------------------------------------------------------
extern "C" void kernel_launch(void* const* d_in, const int* in_sizes, int n_in,
                              void* d_out, int out_size, void* d_ws, size_t ws_size,
                              hipStream_t stream)
{
    const float* ue    = (const float*)d_in[0];
    const float* ie    = (const float*)d_in[1];
    const float* gv[3] = {(const float*)d_in[2], (const float*)d_in[3], (const float*)d_in[4]};
    const int*   users = (const int*)d_in[5];
    const int*   pos   = (const int*)d_in[6];
    const int*   neg   = (const int*)d_in[7];
    const int*   h_idx = (const int*)d_in[8];
    const int*   t_idx = (const int*)d_in[9];
    float*       out   = (float*)d_out;

    // workspace layout (all 16B-aligned offsets):
    //   l1, l2 : full-node layer buffers, NNODES*64 f32 each (64 MB each)
    //   bat[7] : 4096x64 f32 batch mats (ua,pa,na, zu,zuu, zi,zii)
    //   hh[4]  : 4096x64 f16 normalized mats (zu,zuu,zi,zii)
    char*  ws   = (char*)d_ws;
    size_t bufB = (size_t)NNODES * DEMB * sizeof(float);
    size_t batN = (size_t)BATCH * DEMB;
    float* l1   = (float*)ws;
    float* l2   = (float*)(ws + bufB);
    float* bat[7];
    for (int i = 0; i < 7; ++i)
        bat[i] = (float*)(ws + 2 * bufB) + (size_t)i * batN;
    _Float16* hh[4];
    for (int i = 0; i < 4; ++i)
        hh[i] = (_Float16*)(ws + 2 * bufB + 7 * batN * sizeof(float)) + (size_t)i * batN;

    hipMemsetAsync(d_out, 0, (size_t)out_size * sizeof(float), stream);  // svd_loss=0 too

    const dim3 TB(256);
    const int SPMM_BLOCKS = 8192;                      // grid-stride over 64M slots
    const int EXT_BLOCKS  = (BATCH * 16 + 255) / 256;  // 256
    const int ROW_BLOCKS  = (BATCH + 255) / 256;       // 16

    // per val-set: which (index, node_off, dest-batch) to extract
    // set0 -> ua(0), pa(1), na(2); set1 -> zu(3), zi(5); set2 -> zuu(4), zii(6)
    for (int s = 0; s < 3; ++s) {
        hipMemsetAsync(l1, 0, bufB, stream);
        sgl_spmm_kernel<<<SPMM_BLOCKS, TB, 0, stream>>>(h_idx, t_idx, gv[s],
                                                        ue, ie, NUSERS, l1);
        hipMemsetAsync(l2, 0, bufB, stream);
        sgl_spmm_kernel<<<SPMM_BLOCKS, TB, 0, stream>>>(h_idx, t_idx, gv[s],
                                                        l1, l1, NNODES, l2);
        if (s == 0) {
            sgl_extract_kernel<<<EXT_BLOCKS, TB, 0, stream>>>(ue, l1, l2, users, 0,      bat[0]);
            sgl_extract_kernel<<<EXT_BLOCKS, TB, 0, stream>>>(ie, l1, l2, pos,   NUSERS, bat[1]);
            sgl_extract_kernel<<<EXT_BLOCKS, TB, 0, stream>>>(ie, l1, l2, neg,   NUSERS, bat[2]);
        } else if (s == 1) {
            sgl_extract_kernel<<<EXT_BLOCKS, TB, 0, stream>>>(ue, l1, l2, users, 0,      bat[3]);
            sgl_extract_kernel<<<EXT_BLOCKS, TB, 0, stream>>>(ie, l1, l2, pos,   NUSERS, bat[5]);
        } else {
            sgl_extract_kernel<<<EXT_BLOCKS, TB, 0, stream>>>(ue, l1, l2, users, 0,      bat[4]);
            sgl_extract_kernel<<<EXT_BLOCKS, TB, 0, stream>>>(ie, l1, l2, pos,   NUSERS, bat[6]);
        }
    }

    // BPR + embedding regularizer
    sgl_bpr_kernel<<<ROW_BLOCKS, TB, 0, stream>>>(bat[0], bat[1], bat[2], out);
    sgl_emb_kernel<<<ROW_BLOCKS, TB, 0, stream>>>(ue, ie, users, pos, neg, out);

    // normalize the four SSL views to f16
    sgl_norm_f16_kernel<<<ROW_BLOCKS, TB, 0, stream>>>(bat[3], hh[0]);  // zu[users]
    sgl_norm_f16_kernel<<<ROW_BLOCKS, TB, 0, stream>>>(bat[4], hh[1]);  // zuu[users]
    sgl_norm_f16_kernel<<<ROW_BLOCKS, TB, 0, stream>>>(bat[5], hh[2]);  // zi[pos]
    sgl_norm_f16_kernel<<<ROW_BLOCKS, TB, 0, stream>>>(bat[6], hh[3]);  // zii[pos]

    // cse = SSL*( mean(ttl1)-mean(pos1) + mean(ttl2)-mean(pos2) )
    const float ttlScale = SSL_REG / (float)BATCH;
    const float posScale = -SSL_REG / ((float)BATCH * TEMPER);
    sgl_nce_gemm_kernel<<<32, TB, 0, stream>>>(hh[0], hh[1], out + 2, ttlScale);
    sgl_nce_pos_kernel<<<ROW_BLOCKS, TB, 0, stream>>>(hh[0], hh[1], out + 2, posScale);
    sgl_nce_gemm_kernel<<<32, TB, 0, stream>>>(hh[2], hh[3], out + 2, ttlScale);
    sgl_nce_pos_kernel<<<ROW_BLOCKS, TB, 0, stream>>>(hh[2], hh[3], out + 2, posScale);
}
